// NystromSDPA_71451075937038
// MI455X (gfx1250) — matrix operand order repaired and checked by reading, not compiled, not run on hardware
//
#include <hip/hip_runtime.h>
#include <hip/hip_bf16.h>
#include <math.h>

typedef __attribute__((ext_vector_type(16))) _Float16 v16h;
typedef __attribute__((ext_vector_type(8)))  float    v8f;
typedef __attribute__((ext_vector_type(2)))  float    v2f;

#define BATCH 4
#define SEQ   4096
#define HEADS 8
#define DIM   64
#define LM    256          // landmarks
#define LSEG  16           // rows averaged per landmark
#define BH    (BATCH*HEADS)
#define KW    33           // conv kernel
#define SCALE 0.125f       // 64^-0.5

// ---------------------------------------------------------------------------
// K1: landmarks.  q_land = scale * mean of 16 consecutive rows, k_land = mean.
// ---------------------------------------------------------------------------
__global__ void k_landmarks(const float* __restrict__ q, const float* __restrict__ k,
                            float* __restrict__ qland, float* __restrict__ kland) {
    int i  = blockIdx.x;            // landmark index 0..255
    int bh = blockIdx.y;            // 0..31
    int b = bh >> 3, h = bh & 7;
    int dd = threadIdx.x;           // 0..63
    size_t base = ((size_t)(b*SEQ + i*LSEG)*HEADS + h)*DIM + dd;
    float sq = 0.f, sk = 0.f;
    #pragma unroll
    for (int j = 0; j < LSEG; ++j) {
        sq += q[base + (size_t)j*HEADS*DIM];
        sk += k[base + (size_t)j*HEADS*DIM];
    }
    qland[((size_t)bh*LM + i)*DIM + dd] = sq * (SCALE / (float)LSEG);
    kland[((size_t)bh*LM + i)*DIM + dd] = sk * (1.f / (float)LSEG);
}

// ---------------------------------------------------------------------------
// K2: attn2 = softmax(q_land @ k_land^T), 256x256 per bh.  f32 WMMA 16x16x4.
// grid = (16 row-tiles, BH), block = 256 (8 waves); each wave does 2 col tiles.
// ---------------------------------------------------------------------------
__global__ void k_attn2_softmax(const float* __restrict__ qland,
                                const float* __restrict__ kland,
                                float* __restrict__ x) {
    int bh = blockIdx.y;
    int i0 = blockIdx.x * 16;
    int tid = threadIdx.x, lane = tid & 31, wv = tid >> 5;
    const float* A  = qland + (size_t)bh*LM*DIM;
    const float* Bm = kland + (size_t)bh*LM*DIM;
    __shared__ float lg[16][LM];
    __shared__ float red[16][16];

    int arow = i0 + (lane & 15);
    int sel  = (lane & 16) ? 2 : 0;
    for (int tt = 0; tt < 2; ++tt) {
        int j0 = (wv*2 + tt) * 16;
        int bcol = j0 + (lane & 15);
        v8f acc = {};
        #pragma unroll
        for (int kb = 0; kb < DIM; kb += 4) {
            v2f af, bf;
            af[0] = A[(size_t)arow*DIM + kb + sel];
            af[1] = A[(size_t)arow*DIM + kb + sel + 1];
            bf[0] = Bm[(size_t)bcol*DIM + kb + sel];
            bf[1] = Bm[(size_t)bcol*DIM + kb + sel + 1];
            acc = __builtin_amdgcn_wmma_f32_16x16x4_f32(false, af, false, bf,
                                                        (short)0, acc, false, false);
        }
        int rb = (lane & 16) ? 8 : 0;
        #pragma unroll
        for (int r = 0; r < 8; ++r) lg[rb + r][j0 + (lane & 15)] = acc[r];
    }
    __syncthreads();

    int row = tid >> 4, c = tid & 15;
    float mx = -1e30f;
    for (int j = c; j < LM; j += 16) mx = fmaxf(mx, lg[row][j]);
    red[row][c] = mx; __syncthreads();
    if (c == 0) { float m2 = red[row][0];
        for (int t = 1; t < 16; ++t) m2 = fmaxf(m2, red[row][t]);
        red[row][0] = m2; }
    __syncthreads();
    float rm = red[row][0];
    float s = 0.f;
    for (int j = c; j < LM; j += 16) { float e = __expf(lg[row][j] - rm); lg[row][j] = e; s += e; }
    __syncthreads();
    red[row][c] = s; __syncthreads();
    if (c == 0) { float t2 = 0.f;
        for (int t = 0; t < 16; ++t) t2 += red[row][t];
        red[row][0] = t2; }
    __syncthreads();
    float inv = 1.f / red[row][0];
    for (int j = c; j < LM; j += 16)
        x[((size_t)bh*LM + i0 + row)*LM + j] = lg[row][j] * inv;
}

// ---------------------------------------------------------------------------
// K3: pinv init. rows of x sum to 1 => max row-sum term is 1;
// z0 = x^T / max_col_sum.
// ---------------------------------------------------------------------------
__global__ void k_pinv_init(const float* __restrict__ x, float* __restrict__ z) {
    int bh = blockIdx.x, j = threadIdx.x;
    const float* xb = x + (size_t)bh*LM*LM;
    float s = 0.f;
    for (int i = 0; i < LM; ++i) s += fabsf(xb[(size_t)i*LM + j]);
    __shared__ float red[LM];
    red[j] = s; __syncthreads();
    for (int st = LM/2; st > 0; st >>= 1) {
        if (j < st) red[j] = fmaxf(red[j], red[j + st]);
        __syncthreads();
    }
    float inv = 1.f / red[0];
    float* zb = z + (size_t)bh*LM*LM;
    for (int i = 0; i < LM; ++i) zb[(size_t)j*LM + i] = xb[(size_t)i*LM + j] * inv;
}

// ---------------------------------------------------------------------------
// K4a: square-case LDS-tiled matmul:  C = c0*(A@B) + c1*I + c2*A,
// A,B,C: 256x256 per bh.  Block = 256 thr (8 waves) computing a 32x128 tile;
// waves arranged 2x4, each wave: 16 rows x 32 cols (2 accumulators).
// A(32x32) and B(32x128) chunks staged through LDS; frags fed by ds_load.
// grid = (16, BH):  blockIdx.x = rowTile*2 + colTile  (8 x 2).
// ---------------------------------------------------------------------------
__global__ void k_mm256(const float* __restrict__ A, const float* __restrict__ Bm,
                        float* __restrict__ C,
                        float c0, float c1, float c2) {
    int bh  = blockIdx.y;
    int tid = threadIdx.x, lane = tid & 31, wv = tid >> 5;
    int ti0 = (blockIdx.x >> 1) * 32;       // 8 row groups of 32
    int tj0 = (blockIdx.x & 1) * 128;       // 2 col groups of 128
    int wr  = wv >> 2;                      // 0..1  (16-row slice)
    int wc  = wv & 3;                       // 0..3  (32-col slice)
    const float* Ab = A  + (size_t)bh*LM*LM;
    const float* Bb = Bm + (size_t)bh*LM*LM;
    float*       Cb = C  + (size_t)bh*LM*LM;

    __shared__ float As[32][36];            // +4 pad: row-strided reads conflict-free
    __shared__ float Bs[32][132];

    int arow = wr*16 + (lane & 15);         // local A row for this lane
    int sel  = (lane & 16) ? 2 : 0;
    int bc0  = wc*32 + (lane & 15);
    int bc1  = bc0 + 16;

    v8f acc0 = {}, acc1 = {};
    for (int kb = 0; kb < LM; kb += 32) {
        __syncthreads();                    // previous chunk consumed
        // stage A chunk: 32 rows x 32 k  (4 floats/thread)
        {
            int r  = tid >> 3;
            int c4 = (tid & 7) * 4;
            #pragma unroll
            for (int i = 0; i < 4; ++i)
                As[r][c4 + i] = Ab[(size_t)(ti0 + r)*LM + kb + c4 + i];
        }
        // stage B chunk: 32 k x 128 cols (16 floats/thread)
        #pragma unroll
        for (int g = 0; g < 4; ++g) {
            int kr = g*8 + (tid >> 5);
            int cc = (tid & 31) * 4;
            #pragma unroll
            for (int i = 0; i < 4; ++i)
                Bs[kr][cc + i] = Bb[(size_t)(kb + kr)*LM + tj0 + cc + i];
            // prefetch next chunk's B rows into cache (global_prefetch_b8)
            if (kb + 32 < LM)
                __builtin_prefetch(&Bb[(size_t)(kb + 32 + kr)*LM + tj0 + cc], 0, 3);
        }
        __syncthreads();
        // consume: 8 K-steps of 4, two col tiles per wave
        #pragma unroll
        for (int kk = 0; kk < 32; kk += 4) {
            v2f af, bf0, bf1;
            af[0]  = As[arow][kk + sel];
            af[1]  = As[arow][kk + sel + 1];
            bf0[0] = Bs[kk + sel    ][bc0];
            bf0[1] = Bs[kk + sel + 1][bc0];
            bf1[0] = Bs[kk + sel    ][bc1];
            bf1[1] = Bs[kk + sel + 1][bc1];
            acc0 = __builtin_amdgcn_wmma_f32_16x16x4_f32(false, af, false, bf0,
                                                         (short)0, acc0, false, false);
            acc1 = __builtin_amdgcn_wmma_f32_16x16x4_f32(false, af, false, bf1,
                                                         (short)0, acc1, false, false);
        }
    }

    int rb = (lane & 16) ? 8 : 0;
    int gcol0 = tj0 + bc0, gcol1 = tj0 + bc1;
    #pragma unroll
    for (int r = 0; r < 8; ++r) {
        int grow = ti0 + wr*16 + rb + r;
        float v0 = c0 * acc0[r];
        float v1 = c0 * acc1[r];
        if (grow == gcol0) v0 += c1;
        if (grow == gcol1) v1 += c1;
        if (c2 != 0.f) {
            v0 += c2 * Ab[(size_t)grow*LM + gcol0];
            v1 += c2 * Ab[(size_t)grow*LM + gcol1];
        }
        Cb[(size_t)grow*LM + gcol0] = v0;
        Cb[(size_t)grow*LM + gcol1] = v1;
    }
}

// ---------------------------------------------------------------------------
// K4b: rectangular matmul C = c0*(A@B), A:256x256, B:256x64 (single launch,
// direct-from-global operands; traffic is tiny).
// ---------------------------------------------------------------------------
__global__ void k_mm_rect(const float* __restrict__ A, const float* __restrict__ Bm,
                          float* __restrict__ C, int ncols, float c0) {
    int bh = blockIdx.y;
    int tid = threadIdx.x, lane = tid & 31, wv = tid >> 5;
    int strip = blockIdx.x * 8 + wv;            // strips of 16x32
    int nstrips_j = ncols >> 5;
    int ti = strip / nstrips_j;
    int tj = (strip % nstrips_j) * 2;
    const float* Ab = A  + (size_t)bh*LM*LM;
    const float* Bb = Bm + (size_t)bh*LM*ncols;
    float*       Cb = C  + (size_t)bh*LM*ncols;

    int arow = ti*16 + (lane & 15);
    int sel  = (lane & 16) ? 2 : 0;
    int bcol0 = (tj+0)*16 + (lane & 15);
    int bcol1 = (tj+1)*16 + (lane & 15);
    v8f acc0 = {}, acc1 = {};
    #pragma unroll 4
    for (int kb = 0; kb < LM; kb += 4) {
        v2f af, bf0, bf1;
        af[0]  = Ab[(size_t)arow*LM + kb + sel];
        af[1]  = Ab[(size_t)arow*LM + kb + sel + 1];
        bf0[0] = Bb[(size_t)(kb + sel    )*ncols + bcol0];
        bf0[1] = Bb[(size_t)(kb + sel + 1)*ncols + bcol0];
        bf1[0] = Bb[(size_t)(kb + sel    )*ncols + bcol1];
        bf1[1] = Bb[(size_t)(kb + sel + 1)*ncols + bcol1];
        acc0 = __builtin_amdgcn_wmma_f32_16x16x4_f32(false, af, false, bf0,
                                                     (short)0, acc0, false, false);
        acc1 = __builtin_amdgcn_wmma_f32_16x16x4_f32(false, af, false, bf1,
                                                     (short)0, acc1, false, false);
    }
    int rb = (lane & 16) ? 8 : 0;
    #pragma unroll
    for (int r = 0; r < 8; ++r) {
        int row = ti*16 + rb + r;
        Cb[(size_t)row*ncols + bcol0] = c0 * acc0[r];
        Cb[(size_t)row*ncols + bcol1] = c0 * acc1[r];
    }
}

// ---------------------------------------------------------------------------
// K5: kv = softmax(q_land @ k^T) @ v   (flash-style online softmax over n).
// grid = (16 row-tiles, BH), block = 128 (4 waves). Chunk = 64 keys/iter.
// ---------------------------------------------------------------------------
__global__ void k_flashkv(const float* __restrict__ qland,
                          const float* __restrict__ kin,
                          const float* __restrict__ vin,
                          float* __restrict__ kv) {
    int bh = blockIdx.y; int b = bh >> 3, h = bh & 7;
    int i0 = blockIdx.x * 16;
    int tid = threadIdx.x, lane = tid & 31, wv = tid >> 5;   // 4 waves

    __shared__ _Float16 Pbuf[16][64];
    __shared__ float pmax[4][16], psum[4][16];
    __shared__ float mrow[16], srow[16], fac[16], mnewS[16];
    if (tid < 16) { mrow[tid] = -1e30f; srow[tid] = 0.f; }

    const float* Aq = qland + (size_t)bh*LM*DIM;
    v16h a0, a1;
    {
        int mr = i0 + (lane & 15);
        int ks = (lane & 16) ? 8 : 0;
        #pragma unroll
        for (int t = 0; t < 16; ++t) {
            int kk = (t & 7) + ((t >> 3) * 16) + ks;
            a0[t] = (_Float16)Aq[(size_t)mr*DIM + kk];
            a1[t] = (_Float16)Aq[(size_t)mr*DIM + 32 + kk];
        }
    }
    v8f O = {};
    __syncthreads();

    for (int cb = 0; cb < SEQ; cb += 64) {
        int jb = cb + wv*16;
        v16h b0, b1;
        {
            int key = jb + (lane & 15);
            int ks  = (lane & 16) ? 8 : 0;
            const float* kp = kin + ((size_t)(b*SEQ + key)*HEADS + h)*DIM;
            #pragma unroll
            for (int t = 0; t < 16; ++t) {
                b0[t] = (_Float16)kp[(t & 7) + ((t >> 3) * 16) + ks];
                b1[t] = (_Float16)kp[32 + (t & 7) + ((t >> 3) * 16) + ks];
            }
        }
        v8f lt = {};
        lt = __builtin_amdgcn_wmma_f32_16x16x32_f16(false, a0, false, b0, (short)0, lt, false, false);
        lt = __builtin_amdgcn_wmma_f32_16x16x32_f16(false, a1, false, b1, (short)0, lt, false, false);

        int rb = (lane & 16) ? 8 : 0;
        float rm[8];
        #pragma unroll
        for (int r = 0; r < 8; ++r) {
            float v = lt[r];
            #pragma unroll
            for (int m = 1; m < 16; m <<= 1) v = fmaxf(v, __shfl_xor(v, m, 32));
            rm[r] = v;
        }
        if ((lane & 15) == 0)
            for (int r = 0; r < 8; ++r) pmax[wv][rb + r] = rm[r];
        __syncthreads();
        if (tid < 16) {
            float mn = mrow[tid];
            for (int w2 = 0; w2 < 4; ++w2) mn = fmaxf(mn, pmax[w2][tid]);
            fac[tid]   = __expf(mrow[tid] - mn);
            mnewS[tid] = mn;
            mrow[tid]  = mn;
        }
        __syncthreads();

        float ps[8];
        #pragma unroll
        for (int r = 0; r < 8; ++r) {
            float p = __expf(lt[r] - mnewS[rb + r]);
            Pbuf[rb + r][wv*16 + (lane & 15)] = (_Float16)p;
            float v = p;
            #pragma unroll
            for (int m = 1; m < 16; m <<= 1) v += __shfl_xor(v, m, 32);
            ps[r] = v;
            O[r] *= fac[rb + r];
        }
        if ((lane & 15) == 0)
            for (int r = 0; r < 8; ++r) psum[wv][rb + r] = ps[r];
        __syncthreads();
        if (tid < 16) {
            float s = srow[tid] * fac[tid];
            for (int w2 = 0; w2 < 4; ++w2) s += psum[w2][tid];
            srow[tid] = s;
        }

        v16h pa0, pa1, vb0, vb1;
        {
            int mr = (lane & 15);
            int ks = (lane & 16) ? 8 : 0;
            #pragma unroll
            for (int t = 0; t < 16; ++t) {
                int kk = (t & 7) + ((t >> 3) * 16) + ks;
                pa0[t] = Pbuf[mr][kk];
                pa1[t] = Pbuf[mr][32 + kk];
            }
            int colv = wv*16 + (lane & 15);
            int ks2  = (lane & 16) ? 8 : 0;
            #pragma unroll
            for (int t = 0; t < 16; ++t) {
                int key0 = cb + (t & 7) + ((t >> 3) * 16) + ks2;
                vb0[t] = (_Float16)vin[((size_t)(b*SEQ + key0     )*HEADS + h)*DIM + colv];
                vb1[t] = (_Float16)vin[((size_t)(b*SEQ + key0 + 32)*HEADS + h)*DIM + colv];
            }
        }
        O = __builtin_amdgcn_wmma_f32_16x16x32_f16(false, pa0, false, vb0, (short)0, O, false, false);
        O = __builtin_amdgcn_wmma_f32_16x16x32_f16(false, pa1, false, vb1, (short)0, O, false, false);
        __syncthreads();
    }

    __syncthreads();
    int rb = (lane & 16) ? 8 : 0;
    int colv = wv*16 + (lane & 15);
    float* kvb = kv + (size_t)bh*LM*DIM;
    #pragma unroll
    for (int r = 0; r < 8; ++r)
        kvb[(size_t)(i0 + rb + r)*DIM + colv] = O[r] / srow[rb + r];
}

// ---------------------------------------------------------------------------
// K7: out tile = softmax(q_row_tile @ k_land^T) @ w  + depthwise conv residual
// grid = (256 row-tiles, BH), block = 256 (8 waves).
// ---------------------------------------------------------------------------
__global__ void k_final(const float* __restrict__ qin,
                        const float* __restrict__ kland,
                        const float* __restrict__ w,
                        const float* __restrict__ vin,
                        const float* __restrict__ resw,
                        float* __restrict__ out) {
    int bh = blockIdx.y; int b = bh >> 3, h = bh & 7;
    int i0 = blockIdx.x * 16;
    int tid = threadIdx.x, lane = tid & 31, wv = tid >> 5;

    __shared__ float    lg[16][LM];
    __shared__ _Float16 Pb[16][LM];
    __shared__ float    red[16][16];
    __shared__ float    resb[16][DIM];
    __shared__ float    rw[KW];
    if (tid < KW) rw[tid] = resw[h*KW + tid];

    v16h a0, a1;
    {
        int mr = i0 + (lane & 15);
        int ks = (lane & 16) ? 8 : 0;
        const float* qp = qin + ((size_t)(b*SEQ + mr)*HEADS + h)*DIM;
        #pragma unroll
        for (int t = 0; t < 16; ++t) {
            int kk = (t & 7) + ((t >> 3) * 16) + ks;
            a0[t] = (_Float16)(qp[kk]      * SCALE);
            a1[t] = (_Float16)(qp[32 + kk] * SCALE);
        }
    }
    const float* kl = kland + (size_t)bh*LM*DIM;
    for (int tt = 0; tt < 2; ++tt) {
        int j0 = (wv*2 + tt) * 16;
        int jc = j0 + (lane & 15);
        int ks = (lane & 16) ? 8 : 0;
        v16h b0, b1;
        #pragma unroll
        for (int t = 0; t < 16; ++t) {
            b0[t] = (_Float16)kl[(size_t)jc*DIM + (t & 7) + ((t >> 3) * 16) + ks];
            b1[t] = (_Float16)kl[(size_t)jc*DIM + 32 + (t & 7) + ((t >> 3) * 16) + ks];
        }
        v8f lt = {};
        lt = __builtin_amdgcn_wmma_f32_16x16x32_f16(false, a0, false, b0, (short)0, lt, false, false);
        lt = __builtin_amdgcn_wmma_f32_16x16x32_f16(false, a1, false, b1, (short)0, lt, false, false);
        int rb = (lane & 16) ? 8 : 0;
        #pragma unroll
        for (int r = 0; r < 8; ++r) lg[rb + r][j0 + (lane & 15)] = lt[r];
    }
    __syncthreads();

    int row = tid >> 4, c = tid & 15;
    float mx = -1e30f;
    for (int j = c; j < LM; j += 16) mx = fmaxf(mx, lg[row][j]);
    red[row][c] = mx; __syncthreads();
    if (c == 0) { float m2 = red[row][0];
        for (int t = 1; t < 16; ++t) m2 = fmaxf(m2, red[row][t]);
        red[row][0] = m2; }
    __syncthreads();
    float rm = red[row][0];
    float s = 0.f;
    for (int j = c; j < LM; j += 16) { float e = __expf(lg[row][j] - rm); lg[row][j] = e; s += e; }
    __syncthreads();
    red[row][c] = s; __syncthreads();
    if (c == 0) { float t2 = 0.f;
        for (int t = 0; t < 16; ++t) t2 += red[row][t];
        red[row][0] = t2; }
    __syncthreads();
    float inv = 1.f / red[row][0];
    for (int j = c; j < LM; j += 16) Pb[row][j] = (_Float16)(lg[row][j] * inv);

    for (int e = tid; e < 16*DIM; e += 256) {
        int rr = e >> 6, cc = e & 63;
        int srw = i0 + rr;
        float acc = 0.f;
        #pragma unroll
        for (int t = 0; t < KW; ++t) {
            int sr = srw + t - (KW/2);
            if (sr >= 0 && sr < SEQ)
                acc += rw[t] * vin[((size_t)(b*SEQ + sr)*HEADS + h)*DIM + cc];
        }
        resb[rr][cc] = acc;
    }
    __syncthreads();

    if (wv < 4) {
        const float* wb = w + (size_t)bh*LM*DIM;
        int colv = wv*16 + (lane & 15);
        v8f O = {};
        #pragma unroll 2
        for (int kb = 0; kb < LM; kb += 32) {
            v16h pa, wf;
            int mr = (lane & 15);
            int ks = (lane & 16) ? 8 : 0;
            #pragma unroll
            for (int t = 0; t < 16; ++t) {
                int kk = (t & 7) + ((t >> 3) * 16) + ks;
                pa[t] = Pb[mr][kb + kk];
            }
            int ks2 = (lane & 16) ? 8 : 0;
            #pragma unroll
            for (int t = 0; t < 16; ++t)
                wf[t] = (_Float16)wb[(size_t)(kb + (t & 7) + ((t >> 3) * 16) + ks2)*DIM + colv];
            O = __builtin_amdgcn_wmma_f32_16x16x32_f16(false, pa, false, wf, (short)0, O, false, false);
        }
        int rb = (lane & 16) ? 8 : 0;
        #pragma unroll
        for (int r = 0; r < 8; ++r) {
            int orow = i0 + rb + r;
            out[((size_t)(b*SEQ + orow)*HEADS + h)*DIM + colv] = O[r] + resb[rb + r][colv];
        }
    }
}

// ---------------------------------------------------------------------------
extern "C" void kernel_launch(void* const* d_in, const int* in_sizes, int n_in,
                              void* d_out, int out_size, void* d_ws, size_t ws_size,
                              hipStream_t stream) {
    const float* q  = (const float*)d_in[0];
    const float* k  = (const float*)d_in[1];
    const float* v  = (const float*)d_in[2];
    const float* rw = (const float*)d_in[3];
    float* out = (float*)d_out;

    const size_t S_LD = (size_t)BH*LM*DIM;   // 524288 floats
    const size_t S_MM = (size_t)BH*LM*LM;    // 2097152 floats
    float* ws    = (float*)d_ws;
    float* qland = ws;              ws += S_LD;
    float* kland = ws;              ws += S_LD;
    float* x     = ws;              ws += S_MM;
    float* za    = ws;              ws += S_MM;
    float* zb    = ws;              ws += S_MM;
    float* xz    = ws;              ws += S_MM;
    float* t1    = ws;              ws += S_MM;
    float* t2    = ws;              ws += S_MM;
    float* kv    = ws;              ws += S_LD;
    float* wmat  = ws;              ws += S_LD;
    (void)in_sizes; (void)n_in; (void)out_size; (void)ws_size;

    k_landmarks<<<dim3(LM, BH), 64, 0, stream>>>(q, k, qland, kland);
    k_attn2_softmax<<<dim3(16, BH), 256, 0, stream>>>(qland, kland, x);
    k_pinv_init<<<BH, LM, 0, stream>>>(x, za);

    float* zc = za; float* zn = zb;
    for (int it = 0; it < 6; ++it) {
        // xz = x @ z
        k_mm256<<<dim3(16, BH), 256, 0, stream>>>(x,  zc, xz, 1.f,   0.f,  0.f);
        // t1 = 15I - xz@(7I - xz) = xz@xz + 15I - 7*xz
        k_mm256<<<dim3(16, BH), 256, 0, stream>>>(xz, xz, t1, 1.f,  15.f, -7.f);
        // t2 = 13I - xz@t1
        k_mm256<<<dim3(16, BH), 256, 0, stream>>>(xz, t1, t2, -1.f, 13.f,  0.f);
        // z' = 0.25 * z @ t2
        k_mm256<<<dim3(16, BH), 256, 0, stream>>>(zc, t2, zn, 0.25f, 0.f,  0.f);
        float* tmp = zc; zc = zn; zn = tmp;
    }

    // kv = softmax(q_land @ k^T) @ v
    k_flashkv<<<dim3(16, BH), 128, 0, stream>>>(qland, k, v, kv);
    // w = pinv @ kv
    k_mm_rect<<<dim3(4, BH), 256, 0, stream>>>(zc, kv, wmat, DIM, 1.f);
    // out = softmax(q@k_land^T) @ w + conv residual
    k_final<<<dim3(SEQ/16, BH), 256, 0, stream>>>(q, kland, wmat, v, rw, out);
}
